// Attention_91225105367053
// MI455X (gfx1250) — compile-verified, matrix operand order checked by reading, not run on hardware
//
#include <hip/hip_runtime.h>
#include <math.h>

#define BATCH  32
#define SEQ    2048
#define DHID   512
#define ENC2   1024          // 2*ENC_HID
#define INDIM  1536          // ENC2 + DHID

typedef __attribute__((ext_vector_type(16))) __bf16        v16bf;
typedef __attribute__((ext_vector_type(8)))  float         v8f;
typedef __attribute__((ext_vector_type(8)))  unsigned int  v8u;

union Frag {
    v8u   u32;   // 8 dwords = 16 packed bf16
    v16bf bf;
};

// Hardware tanh (v_tanh_f32) if the builtin exists, else libm fallback.
#if __has_builtin(__builtin_amdgcn_tanh_f32)
  #define TANHF(x) __builtin_amdgcn_tanh_f32(x)
#elif __has_builtin(__builtin_amdgcn_tanhf)
  #define TANHF(x) __builtin_amdgcn_tanhf(x)
#else
  #define TANHF(x) tanhf(x)
#endif

// Pack two f32 into packed bf16 (truncation) with ONE v_perm_b32:
// dst = { hi[31:16], lo[31:16] }.  sel bytes (3..0) = (7,6,3,2).
__device__ __forceinline__ unsigned int pack_bf2(float lo, float hi) {
    return __builtin_amdgcn_perm(__builtin_bit_cast(unsigned int, hi),
                                 __builtin_bit_cast(unsigned int, lo),
                                 0x07060302u);
}

// ---------------------------------------------------------------------------
// K0: Webf[o][k] = bf16(W[o][512+k])  -- one dword (2 elems) per thread
// ---------------------------------------------------------------------------
__global__ __launch_bounds__(256) void wconv_kernel(
    const float* __restrict__ W, unsigned int* __restrict__ Webf)
{
    int idx = blockIdx.x * 256 + threadIdx.x;   // 0 .. 512*512-1 (dwords)
    int o  = idx >> 9;                          // row
    int kp = idx & 511;                         // dword within row
    float2 f = *(const float2*)(W + (size_t)o * INDIM + DHID + 2 * kp);
    Webf[(size_t)o * 512 + kp] = pack_bf2(f.x, f.y);
}

// ---------------------------------------------------------------------------
// K1: hb[b][o] = bias[o] + hidden[b,:] . W[o, 0:512]        (32x512 outputs)
// ---------------------------------------------------------------------------
__global__ __launch_bounds__(256) void hb_kernel(
    const float* __restrict__ hidden, const float* __restrict__ W,
    const float* __restrict__ bias,   float* __restrict__ hb)
{
    int idx = blockIdx.x * 256 + threadIdx.x;      // 0 .. 32*512-1
    int b = idx >> 9;
    int o = idx & 511;
    const float* h = hidden + b * DHID;
    const float* w = W + (size_t)o * INDIM;        // W_h row o
    float acc = bias[o];
    #pragma unroll 4
    for (int d = 0; d < DHID; d += 4) {
        float4 hv = *(const float4*)(h + d);
        float4 wv = *(const float4*)(w + d);
        acc += hv.x * wv.x + hv.y * wv.y + hv.z * wv.z + hv.w * wv.w;
    }
    hb[idx] = acc;
}

// ---------------------------------------------------------------------------
// K2: per (batch b, 64 seq rows): e_proj via bf16 WMMA over K=1024, fused
//     tanh + dot-with-v  ->  scores[b][s].
// Block: 256 thr = 8 waves. Wave w owns N strip [w*64, w*64+64) (4 B frags,
// loaded once per K-step, reused across 4 M sub-tiles -> 16 WMMAs/K-step).
// ---------------------------------------------------------------------------
__global__ __launch_bounds__(256) void attn_kernel(
    const float* __restrict__ enc,            // (SEQ, BATCH, ENC2) f32
    const unsigned int* __restrict__ Webf,    // (DHID, ENC2) bf16, dword-packed
    const float* __restrict__ hb,             // (BATCH, DHID)
    const float* __restrict__ vvec,           // (DHID)
    float* __restrict__ scores)               // (BATCH, SEQ)
{
    // ping-pong A tile: 64 rows x 32 k bf16, dword-packed (16 dwords/row)
    __shared__ __align__(16) unsigned int Alds[2][64 * 16];
    __shared__ float smScore[64];

    const int s0   = blockIdx.x * 64;
    const int b    = blockIdx.y;
    const int tid  = threadIdx.x;
    const int lane = tid & 31;
    const int half = lane >> 4;          // 0 or 1
    const int lq   = lane & 15;          // 0..15
    const int n0   = (tid >> 5) * 64;    // wave's N strip base

    if (tid < 64) smScore[tid] = 0.0f;

    v8f acc[4][4] = {};                  // [m-subtile][n-frag]

    // A staging: 1024 dwords/K-step, 4 per thread (rows am+16*p, dword ak)
    const int am = tid >> 4;             // 0..15
    const int ak = tid & 15;             // dword index in row (2 k's)
    const float* aptr = enc + (size_t)b * ENC2
                      + (size_t)(s0 + am) * (BATCH * ENC2) + 2 * ak;
    const size_t rowStride16 = (size_t)16 * BATCH * ENC2;   // 16 seq rows

    // B base for this lane (row n0+lq of bf16 W_e); frags at +f*16 rows
    const unsigned int* bbase = Webf + (size_t)(n0 + lq) * 512;

    for (int k0 = 0; k0 < ENC2; k0 += 32) {
        const int buf = (k0 >> 5) & 1;
        // --- stage A tile: 4x (global_load_b64 + v_perm + ds_store_b32)
        #pragma unroll
        for (int p = 0; p < 4; ++p) {
            const float* ap = aptr + p * rowStride16 + k0;
            float2 f2 = *(const float2*)ap;
            __builtin_prefetch(ap + 64, 0, 1);   // next-next K chunk
            Alds[buf][(am + 16 * p) * 16 + ak] = pack_bf2(f2.x, f2.y);
        }
        __syncthreads();

        // --- 4 B fragments (held across all 4 M sub-tiles)
        Frag bm[4];
        #pragma unroll
        for (int f = 0; f < 4; ++f) {
            const uint4* bq = (const uint4*)(bbase + (size_t)f * 16 * 512 + (k0 >> 1));
            uint4 b0 = bq[half];
            uint4 b1 = bq[2 + half];
            bm[f].u32[0] = b0.x; bm[f].u32[1] = b0.y;
            bm[f].u32[2] = b0.z; bm[f].u32[3] = b0.w;
            bm[f].u32[4] = b1.x; bm[f].u32[5] = b1.y;
            bm[f].u32[6] = b1.z; bm[f].u32[7] = b1.w;
        }

        // --- 4 M sub-tiles x 4 N frags = 16 WMMAs
        #pragma unroll
        for (int mt = 0; mt < 4; ++mt) {
            // A fragment per ISA layout: lane M=lq (+16*mt tile row offset);
            // VGPRs 0-3 <-> K=8*half+0..7, VGPRs 4-7 <-> K=16+8*half+0..7
            Frag a;
            const uint4* rowq = (const uint4*)&Alds[buf][(mt * 16 + lq) * 16];
            uint4 w0 = rowq[half];
            uint4 w1 = rowq[2 + half];
            a.u32[0] = w0.x; a.u32[1] = w0.y; a.u32[2] = w0.z; a.u32[3] = w0.w;
            a.u32[4] = w1.x; a.u32[5] = w1.y; a.u32[6] = w1.z; a.u32[7] = w1.w;

            #pragma unroll
            for (int f = 0; f < 4; ++f)
                acc[mt][f] = __builtin_amdgcn_wmma_f32_16x16x32_bf16(
                    false, a.bf, false, bm[f].bf, (short)0, acc[mt][f],
                    false, false);
        }
    }

    // --- epilogue: tanh + dot with v, reduce N away.
    // C/D layout: (lane, VGPR r) -> (M = r + 8*half, N = lq) within each frag.
    const float* hbB = hb + b * DHID;
    float cf[4], vf[4];
    #pragma unroll
    for (int f = 0; f < 4; ++f) {
        int o = n0 + f * 16 + lq;
        cf[f] = hbB[o];
        vf[f] = vvec[o];
    }

    #pragma unroll
    for (int mt = 0; mt < 4; ++mt) {
        float local[8];
        #pragma unroll
        for (int r = 0; r < 8; ++r) local[r] = 0.0f;
        #pragma unroll
        for (int f = 0; f < 4; ++f)
            #pragma unroll
            for (int r = 0; r < 8; ++r)
                local[r] += TANHF(acc[mt][f][r] + cf[f]) * vf[f];
        // butterfly-sum across the 16 lanes of each half (stays within half)
        #pragma unroll
        for (int m = 8; m >= 1; m >>= 1)
            #pragma unroll
            for (int r = 0; r < 8; ++r)
                local[r] += __shfl_xor(local[r], m, 32);
        if (lq == 0) {
            #pragma unroll
            for (int r = 0; r < 8; ++r)
                atomicAdd(&smScore[mt * 16 + r + 8 * half], local[r]);
        }
    }
    __syncthreads();
    if (tid < 64)
        scores[(size_t)b * SEQ + s0 + tid] = smScore[tid];
}

// ---------------------------------------------------------------------------
// K3: softmax over seq axis, one block per batch row
// ---------------------------------------------------------------------------
__global__ __launch_bounds__(256) void softmax_kernel(
    const float* __restrict__ scores, float* __restrict__ out)
{
    __shared__ float red[8];
    const int b   = blockIdx.x;
    const int tid = threadIdx.x;
    const float* row = scores + (size_t)b * SEQ;
    float* orow      = out    + (size_t)b * SEQ;

    float mx = -INFINITY;
    for (int s = tid; s < SEQ; s += 256) mx = fmaxf(mx, row[s]);
    #pragma unroll
    for (int m = 16; m >= 1; m >>= 1) mx = fmaxf(mx, __shfl_xor(mx, m, 32));
    if ((tid & 31) == 0) red[tid >> 5] = mx;
    __syncthreads();
    mx = red[0];
    #pragma unroll
    for (int i = 1; i < 8; ++i) mx = fmaxf(mx, red[i]);

    float sum = 0.0f;
    for (int s = tid; s < SEQ; s += 256) {
        float e = __expf(row[s] - mx);
        orow[s] = e;
        sum += e;
    }
    #pragma unroll
    for (int m = 16; m >= 1; m >>= 1) sum += __shfl_xor(sum, m, 32);
    __syncthreads();
    if ((tid & 31) == 0) red[tid >> 5] = sum;
    __syncthreads();
    sum = 0.0f;
    #pragma unroll
    for (int i = 0; i < 8; ++i) sum += red[i];
    float inv = 1.0f / sum;
    for (int s = tid; s < SEQ; s += 256) orow[s] *= inv;
}

// ---------------------------------------------------------------------------
extern "C" void kernel_launch(void* const* d_in, const int* in_sizes, int n_in,
                              void* d_out, int out_size, void* d_ws, size_t ws_size,
                              hipStream_t stream)
{
    const float* hidden = (const float*)d_in[0];   // (32, 512)
    const float* enc    = (const float*)d_in[1];   // (2048, 32, 1024)
    const float* W      = (const float*)d_in[2];   // (512, 1536)
    const float* bias   = (const float*)d_in[3];   // (512)
    const float* v      = (const float*)d_in[4];   // (512)
    float* out = (float*)d_out;                    // (32, 2048)

    // workspace layout: Webf (512*512 dwords = 1MB) | hb (64KB) | scores (256KB)
    unsigned int* Webf = (unsigned int*)d_ws;
    float* hb     = (float*)(Webf + DHID * (ENC2 / 2));
    float* scores = hb + BATCH * DHID;

    wconv_kernel<<<(DHID * ENC2 / 2) / 256, 256, 0, stream>>>(W, Webf);
    hb_kernel<<<(BATCH * DHID) / 256, 256, 0, stream>>>(hidden, W, bias, hb);
    attn_kernel<<<dim3(SEQ / 64, BATCH), 256, 0, stream>>>(enc, Webf, hb, v, scores);
    softmax_kernel<<<BATCH, 256, 0, stream>>>(scores, out);
}
